// SharedMoELayer_82179904242348
// MI455X (gfx1250) — matrix operand order repaired and checked by reading, not compiled, not run on hardware
//
#include <hip/hip_runtime.h>
#include <hip/hip_bf16.h>

typedef __bf16 bf16;
typedef __attribute__((ext_vector_type(16))) __bf16 bf16x16;
typedef __attribute__((ext_vector_type(8)))  float   f32x8;
typedef __attribute__((ext_vector_type(4)))  unsigned int u32x4;

#define T_TOK 8192
#define DDIM  1024
#define HDIM  4096
#define NEXP  8

#define BM 128
#define BN 128
#define BK 32
#define AKP 40     // padded bf16 per A-row in LDS (40*2B = 80B row stride)
#define BNP 132    // padded dwords per B k-pair row in LDS
#define ASZ (BM * AKP)          // bf16 per A buffer
#define BSZ ((BK / 2) * BNP)    // dwords per B buffer

// ---------------- fp32 -> bf16 cast ----------------
__global__ void cast_f32_to_bf16(const float* __restrict__ s, bf16* __restrict__ d, int n4) {
  int i = blockIdx.x * blockDim.x + threadIdx.x;
  if (i >= n4) return;
  const float4 v = ((const float4*)s)[i];
  union { bf16 b[4]; unsigned long long u; } p;
  p.b[0] = (bf16)v.x; p.b[1] = (bf16)v.y; p.b[2] = (bf16)v.z; p.b[3] = (bf16)v.w;
  ((unsigned long long*)d)[i] = p.u;
}

// ---------------- router: softmax + top-2 + renorm + expert lists ----------------
__global__ void router_topk(const float* __restrict__ x, const float* __restrict__ rw,
                            int* __restrict__ cnt, int* __restrict__ idx,
                            float* __restrict__ wgt) {
  const int lane = threadIdx.x & 31;
  const int wave = threadIdx.x >> 5;
  const int t = blockIdx.x * (blockDim.x >> 5) + wave;
  if (t >= T_TOK) return;
  float l[NEXP];
#pragma unroll
  for (int e = 0; e < NEXP; ++e) l[e] = 0.f;
  const float* xr = x + (size_t)t * DDIM;
  for (int j = 0; j < DDIM / 32; ++j) {
    const int c = j * 32 + lane;
    const float xv = xr[c];
    const float* r = rw + c * NEXP;
#pragma unroll
    for (int e = 0; e < NEXP; ++e) l[e] += xv * r[e];
  }
#pragma unroll
  for (int off = 16; off >= 1; off >>= 1)
#pragma unroll
    for (int e = 0; e < NEXP; ++e) l[e] += __shfl_xor(l[e], off, 32);
  if (lane == 0) {
    float m = l[0];
#pragma unroll
    for (int e = 1; e < NEXP; ++e) m = fmaxf(m, l[e]);
    float p[NEXP];
#pragma unroll
    for (int e = 0; e < NEXP; ++e) p[e] = __expf(l[e] - m);
    int i1 = 0;
#pragma unroll
    for (int e = 1; e < NEXP; ++e) if (p[e] > p[i1]) i1 = e;
    const float p1 = p[i1];
    p[i1] = -1.f;
    int i2 = 0;
#pragma unroll
    for (int e = 1; e < NEXP; ++e) if (p[e] > p[i2]) i2 = e;
    const float p2 = p[i2];
    const float inv = 1.f / (p1 + p2);
    int s1 = atomicAdd(&cnt[i1], 1);
    idx[i1 * T_TOK + s1] = t; wgt[i1 * T_TOK + s1] = p1 * inv;
    int s2 = atomicAdd(&cnt[i2], 1);
    idx[i2 * T_TOK + s2] = t; wgt[i2 * T_TOK + s2] = p2 * inv;
  }
}

// v_perm_b32 pack: dword pair {B[k][n], B[k+1][n]} from two row-segments
__device__ __forceinline__ void pack_pairs(const u32x4 q0, const u32x4 q1,
                                           unsigned int d[8]) {
#pragma unroll
  for (int j = 0; j < 4; ++j) {
    d[2 * j]     = __builtin_amdgcn_perm(q1[j], q0[j], 0x05040100u);
    d[2 * j + 1] = __builtin_amdgcn_perm(q1[j], q0[j], 0x07060302u);
  }
}

// ---------------- GEMM1: h = silu(gather(x) @ W1), bf16 out ----------------
template <bool GATHER>
__launch_bounds__(256)
__global__ void moe_gemm1(const bf16* __restrict__ A, const bf16* __restrict__ B,
                          bf16* __restrict__ Hout, const int* __restrict__ cntp,
                          const int* __restrict__ gidx) {
  __shared__ bf16 sA[2 * ASZ];
  __shared__ unsigned int sB[2 * BSZ];
  __shared__ int sIdx[BM];

  const int rows = GATHER ? cntp[0] : T_TOK;
  const int rowBase = blockIdx.x * BM;
  if (rowBase >= rows) return;
  const int nTile = blockIdx.y * BN;
  const int tid = threadIdx.x;

  if (GATHER) {
    if (tid < BM) {
      const int r = rowBase + tid;
      sIdx[tid] = (r < rows) ? gidx[r] : gidx[rows - 1];
    }
    __syncthreads();
  }

  // global->LDS staging coords
  const int ar  = tid >> 1;           // A row 0..127
  const int ac0 = (tid & 1) * 16;     // A col base (bf16)
  const long aRowG = GATHER ? (long)sIdx[ar] : (long)(rowBase + ar);
  const bf16* aG = A + aRowG * DDIM + ac0;
  const int bp  = tid >> 4;           // k-pair 0..15
  const int bn0 = (tid & 15) * 8;     // 8 columns
  const bf16* bG = B + (long)(2 * bp) * HDIM + nTile + bn0;

  // wave tiling: 8 waves = 2(M) x 4(N); wave tile 64x32
  const int lane = tid & 31;
  const int wid  = tid >> 5;
  const int wm = (wid & 1) * 64;
  const int wn = (wid >> 1) * 32;
  const int l15 = lane & 15;
  const int hi  = lane >> 4;

  f32x8 acc[4][2] = {};

  u32x4 ra0, ra1, rb0, rb1;
  ra0 = *(const u32x4*)(aG);
  ra1 = *(const u32x4*)(aG + 8);
  rb0 = *(const u32x4*)(bG);
  rb1 = *(const u32x4*)(bG + HDIM);

  int buf = 0;
  for (int kt = 0; kt < DDIM; kt += BK) {
    bf16* As = sA + buf * ASZ;
    unsigned int* Bs = sB + buf * BSZ;
    *(u32x4*)(&As[ar * AKP + ac0])     = ra0;
    *(u32x4*)(&As[ar * AKP + ac0 + 8]) = ra1;
    {
      unsigned int d[8];
      pack_pairs(rb0, rb1, d);
      unsigned int* bb = &Bs[bp * BNP + bn0];
      *(u32x4*)(bb)     = *(u32x4*)&d[0];
      *(u32x4*)(bb + 4) = *(u32x4*)&d[4];
    }
    __syncthreads();
    if (kt + BK < DDIM) {
      ra0 = *(const u32x4*)(aG + (kt + BK));
      ra1 = *(const u32x4*)(aG + (kt + BK) + 8);
      rb0 = *(const u32x4*)(bG + (long)(kt + BK) * HDIM);
      rb1 = *(const u32x4*)(bG + (long)(kt + BK + 1) * HDIM);
    }
    bf16x16 af[4];
#pragma unroll
    for (int mi = 0; mi < 4; ++mi) {
      const int row = wm + mi * 16 + l15;
      union { u32x4 q[2]; bf16x16 v; } f;
      f.q[0] = *(const u32x4*)(&As[row * AKP + hi * 8]);
      f.q[1] = *(const u32x4*)(&As[row * AKP + 16 + hi * 8]);
      af[mi] = f.v;
    }
    bf16x16 bfr[2];
#pragma unroll
    for (int ni = 0; ni < 2; ++ni) {
      const int nb = wn + ni * 16 + l15;
      union { unsigned int u[8]; bf16x16 v; } f;
#pragma unroll
      for (int v2 = 0; v2 < 8; ++v2) f.u[v2] = Bs[(hi * 4 + v2 + (v2 & 4)) * BNP + nb];
      bfr[ni] = f.v;
    }
#pragma unroll
    for (int mi = 0; mi < 4; ++mi)
#pragma unroll
      for (int ni = 0; ni < 2; ++ni)
        acc[mi][ni] = __builtin_amdgcn_wmma_f32_16x16x32_bf16(
            false, af[mi], false, bfr[ni], (short)0, acc[mi][ni], false, false);
    buf ^= 1;
  }

  // epilogue: SiLU + bf16 store (C layout: vgpr r -> M = r or r+8; lane -> N)
#pragma unroll
  for (int mi = 0; mi < 4; ++mi)
#pragma unroll
    for (int ni = 0; ni < 2; ++ni) {
      const int n = nTile + wn + ni * 16 + l15;
#pragma unroll
      for (int r = 0; r < 8; ++r) {
        const int m = wm + mi * 16 + hi * 8 + r;
        float v = acc[mi][ni][r];
        v = v / (1.f + __expf(-v));
        Hout[(long)(rowBase + m) * HDIM + n] = (bf16)v;
      }
    }
}

// ---------------- GEMM2: out[tok] (+)= w * (h @ W2) ----------------
template <bool WEIGHTED>
__launch_bounds__(256)
__global__ void moe_gemm2(const bf16* __restrict__ Hm, const bf16* __restrict__ B,
                          float* __restrict__ Out, const int* __restrict__ cntp,
                          const int* __restrict__ gidx, const float* __restrict__ gwgt) {
  __shared__ bf16 sA[2 * ASZ];
  __shared__ unsigned int sB[2 * BSZ];
  __shared__ int sIdx[BM];
  __shared__ float sWgt[BM];

  const int rows = WEIGHTED ? cntp[0] : T_TOK;
  const int rowBase = blockIdx.x * BM;
  if (rowBase >= rows) return;
  const int nTile = blockIdx.y * BN;
  const int tid = threadIdx.x;

  if (WEIGHTED) {
    if (tid < BM) {
      const int r = rowBase + tid;
      sIdx[tid] = (r < rows) ? gidx[r] : 0;
      sWgt[tid] = (r < rows) ? gwgt[r] : 0.f;
    }
    __syncthreads();
  }

  const int ar  = tid >> 1;
  const int ac0 = (tid & 1) * 16;
  const bf16* aG = Hm + (long)(rowBase + ar) * HDIM + ac0;
  const int bp  = tid >> 4;
  const int bn0 = (tid & 15) * 8;
  const bf16* bG = B + (long)(2 * bp) * DDIM + nTile + bn0;

  const int lane = tid & 31;
  const int wid  = tid >> 5;
  const int wm = (wid & 1) * 64;
  const int wn = (wid >> 1) * 32;
  const int l15 = lane & 15;
  const int hi  = lane >> 4;

  f32x8 acc[4][2] = {};

  u32x4 ra0, ra1, rb0, rb1;
  ra0 = *(const u32x4*)(aG);
  ra1 = *(const u32x4*)(aG + 8);
  rb0 = *(const u32x4*)(bG);
  rb1 = *(const u32x4*)(bG + DDIM);

  int buf = 0;
  for (int kt = 0; kt < HDIM; kt += BK) {
    bf16* As = sA + buf * ASZ;
    unsigned int* Bs = sB + buf * BSZ;
    *(u32x4*)(&As[ar * AKP + ac0])     = ra0;
    *(u32x4*)(&As[ar * AKP + ac0 + 8]) = ra1;
    {
      unsigned int d[8];
      pack_pairs(rb0, rb1, d);
      unsigned int* bb = &Bs[bp * BNP + bn0];
      *(u32x4*)(bb)     = *(u32x4*)&d[0];
      *(u32x4*)(bb + 4) = *(u32x4*)&d[4];
    }
    __syncthreads();
    if (kt + BK < HDIM) {
      ra0 = *(const u32x4*)(aG + (kt + BK));
      ra1 = *(const u32x4*)(aG + (kt + BK) + 8);
      rb0 = *(const u32x4*)(bG + (long)(kt + BK) * DDIM);
      rb1 = *(const u32x4*)(bG + (long)(kt + BK + 1) * DDIM);
    }
    bf16x16 af[4];
#pragma unroll
    for (int mi = 0; mi < 4; ++mi) {
      const int row = wm + mi * 16 + l15;
      union { u32x4 q[2]; bf16x16 v; } f;
      f.q[0] = *(const u32x4*)(&As[row * AKP + hi * 8]);
      f.q[1] = *(const u32x4*)(&As[row * AKP + 16 + hi * 8]);
      af[mi] = f.v;
    }
    bf16x16 bfr[2];
#pragma unroll
    for (int ni = 0; ni < 2; ++ni) {
      const int nb = wn + ni * 16 + l15;
      union { unsigned int u[8]; bf16x16 v; } f;
#pragma unroll
      for (int v2 = 0; v2 < 8; ++v2) f.u[v2] = Bs[(hi * 4 + v2 + (v2 & 4)) * BNP + nb];
      bfr[ni] = f.v;
    }
#pragma unroll
    for (int mi = 0; mi < 4; ++mi)
#pragma unroll
      for (int ni = 0; ni < 2; ++ni)
        acc[mi][ni] = __builtin_amdgcn_wmma_f32_16x16x32_bf16(
            false, af[mi], false, bfr[ni], (short)0, acc[mi][ni], false, false);
    buf ^= 1;
  }

#pragma unroll
  for (int mi = 0; mi < 4; ++mi)
#pragma unroll
    for (int ni = 0; ni < 2; ++ni) {
      const int n = nTile + wn + ni * 16 + l15;
#pragma unroll
      for (int r = 0; r < 8; ++r) {
        const int m = wm + mi * 16 + hi * 8 + r;
        const int lr = rowBase + m;
        if (lr < rows) {
          const float v = acc[mi][ni][r];
          if (WEIGHTED) {
            atomicAdd(&Out[(long)sIdx[m] * DDIM + n], v * sWgt[m]);
          } else {
            Out[(long)lr * DDIM + n] = v;
          }
        }
      }
    }
}

// ---------------- host launch ----------------
extern "C" void kernel_launch(void* const* d_in, const int* in_sizes, int n_in,
                              void* d_out, int out_size, void* d_ws, size_t ws_size,
                              hipStream_t stream) {
  (void)in_sizes; (void)n_in; (void)out_size; (void)ws_size;
  const float* x   = (const float*)d_in[0];
  const float* rw  = (const float*)d_in[1];
  const float* ew1 = (const float*)d_in[2];
  const float* ew2 = (const float*)d_in[3];
  const float* sw1 = (const float*)d_in[4];
  const float* sw2 = (const float*)d_in[5];
  float* out = (float*)d_out;

  char* wsp = (char*)d_ws;
  auto alloc = [&](size_t bytes) {
    char* p = wsp;
    wsp += (bytes + 255) & ~(size_t)255;
    return p;
  };
  bf16*  xb  = (bf16*)alloc((size_t)T_TOK * DDIM * 2);
  bf16*  w1b = (bf16*)alloc((size_t)NEXP * DDIM * HDIM * 2);
  bf16*  w2b = (bf16*)alloc((size_t)NEXP * HDIM * DDIM * 2);
  bf16*  s1b = (bf16*)alloc((size_t)DDIM * HDIM * 2);
  bf16*  s2b = (bf16*)alloc((size_t)HDIM * DDIM * 2);
  bf16*  hb  = (bf16*)alloc((size_t)T_TOK * HDIM * 2);
  int*   cnt = (int*)alloc(NEXP * sizeof(int));
  int*   idx = (int*)alloc((size_t)NEXP * T_TOK * sizeof(int));
  float* wgt = (float*)alloc((size_t)NEXP * T_TOK * sizeof(float));

  auto cast = [&](const float* s, bf16* d, size_t n) {
    int n4 = (int)(n / 4);
    cast_f32_to_bf16<<<(n4 + 255) / 256, 256, 0, stream>>>(s, d, n4);
  };
  cast(x,   xb,  (size_t)T_TOK * DDIM);
  cast(ew1, w1b, (size_t)NEXP * DDIM * HDIM);
  cast(ew2, w2b, (size_t)NEXP * HDIM * DDIM);
  cast(sw1, s1b, (size_t)DDIM * HDIM);
  cast(sw2, s2b, (size_t)HDIM * DDIM);

  hipMemsetAsync(cnt, 0, NEXP * sizeof(int), stream);
  router_topk<<<T_TOK / 4, 128, 0, stream>>>(x, rw, cnt, idx, wgt);

  // shared expert: writes every element of out (clears poison), no atomics
  moe_gemm1<false><<<dim3(T_TOK / BM, HDIM / BN), 256, 0, stream>>>(
      xb, s1b, hb, nullptr, nullptr);
  moe_gemm2<false><<<dim3(T_TOK / BM, DDIM / BN), 256, 0, stream>>>(
      hb, s2b, out, nullptr, nullptr, nullptr);

  // routed experts, sequential, reuse hb
  for (int e = 0; e < NEXP; ++e) {
    moe_gemm1<true><<<dim3(T_TOK / BM, HDIM / BN), 256, 0, stream>>>(
        xb, w1b + (size_t)e * DDIM * HDIM, hb, cnt + e, idx + e * T_TOK);
    moe_gemm2<true><<<dim3(T_TOK / BM, DDIM / BN), 256, 0, stream>>>(
        hb, w2b + (size_t)e * HDIM * DDIM, out, cnt + e, idx + e * T_TOK,
        wgt + e * T_TOK);
  }
}
